// Speller_81200651698319
// MI455X (gfx1250) — matrix operand order repaired and checked by reading, not compiled, not run on hardware
//
#include <hip/hip_runtime.h>

typedef __bf16 bf16;
typedef __attribute__((ext_vector_type(16))) __bf16 v16bf;
typedef __attribute__((ext_vector_type(8)))  __bf16 v8bf;
typedef __attribute__((ext_vector_type(8)))  float  v8f;

#define B_   32
#define T_   128
#define T1_  129
#define S_   512
#define H_   512
#define E_   512
#define P_   512
#define V_   10000
#define NH_  8
#define HD_  64
#define L_   2
#define G4H  2048
#define M1   (T1_*B_)   /* 4128  */
#define MS   (S_*B_)    /* 16384 */

// ---------------------------------------------------------------------------
// WMMA GEMM:  C[M,N] = alpha * (A[M,K] @ W[N,K]^T + bias[N]) + Cadd[M,N]
// A, W bf16; C written as f32 (Cf) and/or bf16 (Cbf).  M%16==0, K%32==0, N%16==0.
// One wave computes 16(M) x 64(N); A fragment reused 4x.  The K-loop is
// branch-free: ragged-N tiles clamp their weight-row index (in-bounds duplicate
// loads), and only the epilogue stores are guarded (wave-uniform).
// Fragment layouts used here (wave32; A and B in contiguous k order):
//   A 16x32 bf16: lane(m=ln, half=hf) vector elem j = A[m][16*hf+j]
//   B 32x16 bf16: lane(n=ln, half=hf) vector elem j = B[16*hf+j][n]
//   D 16x16 f32 : lane(n=ln, half=hf) elem r        = C[8*hf+r][n]
// ---------------------------------------------------------------------------
__global__ void __launch_bounds__(128) k_gemm(
    const bf16* __restrict__ A, const bf16* __restrict__ W,
    const float* __restrict__ bias, const float* __restrict__ Cadd,
    float* __restrict__ Cf, bf16* __restrict__ Cbf,
    int M, int N, int K, float alpha)
{
  const int wave = threadIdx.x >> 5;
  const int lane = threadIdx.x & 31;
  const int hf   = lane >> 4;     // wave half
  const int ln   = lane & 15;
  const int mTile = blockIdx.y * 4 + wave;
  if (mTile * 16 >= M) return;                    // wave-uniform guard
  const int m0 = mTile * 16;
  const int n0 = blockIdx.x * 64;

  // Clamped weight-row indices: always in-bounds, duplicates for ragged tiles.
  int c0 = n0 + ln;      if (c0 > N - 1) c0 = N - 1;
  int c1 = n0 + 16 + ln; if (c1 > N - 1) c1 = N - 1;
  int c2 = n0 + 32 + ln; if (c2 > N - 1) c2 = N - 1;
  int c3 = n0 + 48 + ln; if (c3 > N - 1) c3 = N - 1;
  const bf16* W0 = W + (size_t)c0 * K;
  const bf16* W1 = W + (size_t)c1 * K;
  const bf16* W2 = W + (size_t)c2 * K;
  const bf16* W3 = W + (size_t)c3 * K;
  const bf16* Arow = A + (size_t)(m0 + ln) * K;   // lanes 0-15 & 16-31: same M rows

  v8f acc0 = {}, acc1 = {}, acc2 = {}, acc3 = {};

  for (int k0 = 0; k0 < K; k0 += 32) {
    union { v16bf v; v8bf h[2]; } a;
    a.h[0] = *(const v8bf*)(Arow + k0 + 16 * hf);       // K = k0+16hf .. +7
    a.h[1] = *(const v8bf*)(Arow + k0 + 16 * hf + 8);   // K = k0+16hf+8 .. +7
    union { v16bf v; v8bf h[2]; } b0, b1, b2, b3;
    b0.h[0] = *(const v8bf*)(W0 + k0 + 16 * hf);
    b0.h[1] = *(const v8bf*)(W0 + k0 + 16 * hf + 8);
    b1.h[0] = *(const v8bf*)(W1 + k0 + 16 * hf);
    b1.h[1] = *(const v8bf*)(W1 + k0 + 16 * hf + 8);
    b2.h[0] = *(const v8bf*)(W2 + k0 + 16 * hf);
    b2.h[1] = *(const v8bf*)(W2 + k0 + 16 * hf + 8);
    b3.h[0] = *(const v8bf*)(W3 + k0 + 16 * hf);
    b3.h[1] = *(const v8bf*)(W3 + k0 + 16 * hf + 8);
    acc0 = __builtin_amdgcn_wmma_f32_16x16x32_bf16(false, a.v, false, b0.v, (short)0, acc0, false, false);
    acc1 = __builtin_amdgcn_wmma_f32_16x16x32_bf16(false, a.v, false, b1.v, (short)0, acc1, false, false);
    acc2 = __builtin_amdgcn_wmma_f32_16x16x32_bf16(false, a.v, false, b2.v, (short)0, acc2, false, false);
    acc3 = __builtin_amdgcn_wmma_f32_16x16x32_bf16(false, a.v, false, b3.v, (short)0, acc3, false, false);
  }

  // Epilogue: per-tile store guards are wave-uniform (outside the hot loop).
  v8f* accs[4] = { &acc0, &acc1, &acc2, &acc3 };
  for (int nt = 0; nt < 4; ++nt) {
    if ((n0 + nt * 16) >= N) break;               // uniform
    const int col = n0 + nt * 16 + ln;
    const float bcol = bias ? bias[col] : 0.0f;
    const v8f av = *accs[nt];
    for (int r = 0; r < 8; ++r) {
      const int row = m0 + hf * 8 + r;
      float val = (av[r] + bcol) * alpha;
      if (Cadd) val += Cadd[(size_t)row * N + col];
      if (Cf)  Cf [(size_t)row * N + col] = val;
      if (Cbf) Cbf[(size_t)row * N + col] = (bf16)val;
    }
  }
}

// ---------------------------------------------------------------------------
__global__ void k_cvt(const float* __restrict__ s, bf16* __restrict__ d, int n) {
  int i = blockIdx.x * 256 + threadIdx.x;
  if (i < n) d[i] = (bf16)s[i];
}

__global__ void k_zero_u32(unsigned* __restrict__ p, int n) {
  int i = blockIdx.x * 256 + threadIdx.x;
  if (i < n) p[i] = 0u;
}

__global__ void k_bias2(const float* __restrict__ a, const float* __restrict__ b,
                        float* __restrict__ o, int n) {
  int i = blockIdx.x * 256 + threadIdx.x;
  if (i < n) o[i] = a[i] + b[i];
}

// x[t*B+b][e] = emb[tok][e], tok = (t==0 ? SOS=1 : inputs[b][t-1])
__global__ void k_embed(const int* __restrict__ inputs, const float* __restrict__ emb,
                        bf16* __restrict__ x) {
  int i = blockIdx.x * 256 + threadIdx.x;
  if (i >= M1 * E_) return;
  int e = i % E_, row = i / E_;
  int t = row / B_, b = row % B_;
  int tok = (t == 0) ? 1 : inputs[b * T_ + (t - 1)];
  x[i] = (bf16)emb[(size_t)tok * E_ + e];
}

// kv[s*B+b][h] = enc[b][s][h] (bf16)
__global__ void k_prep_kv(const float* __restrict__ enc, bf16* __restrict__ kv) {
  int i = blockIdx.x * 256 + threadIdx.x;
  if (i >= MS * H_) return;
  int h = i % H_, row = i / H_;
  int s = row / B_, b = row % B_;
  kv[i] = (bf16)enc[((size_t)b * S_ + s) * H_ + h];
}

// LSTM cell pointwise: gates g[B,4H] (i,f,g,o) -> c, h(bf16), seq_out[t]
__global__ void k_lstm_cell(const float* __restrict__ g, float* __restrict__ c,
                            bf16* __restrict__ hbf, bf16* __restrict__ seq_out, int t) {
  int i = blockIdx.x * 256 + threadIdx.x;
  if (i >= B_ * H_) return;
  int hh = i % H_, b = i / H_;
  const float* gr = g + (size_t)b * G4H;
  float gi = gr[hh], gf = gr[H_ + hh], gg = gr[2 * H_ + hh], go = gr[3 * H_ + hh];
  float si = 1.0f / (1.0f + __expf(-gi));
  float sf = 1.0f / (1.0f + __expf(-gf));
  float so = 1.0f / (1.0f + __expf(-go));
  float cn = sf * c[i] + si * tanhf(gg);
  float hn = so * tanhf(cn);
  c[i] = cn;
  hbf[i] = (bf16)hn;
  seq_out[(size_t)t * B_ * H_ + i] = (bf16)hn;
}

// Fused attention row: grid(NH,B,T1), block 64.  q pre-scaled by HD^-0.5.
__global__ void __launch_bounds__(64) k_attn(const float* __restrict__ q,
                                             const bf16* __restrict__ k,
                                             const bf16* __restrict__ v,
                                             bf16* __restrict__ ctx) {
  const int h = blockIdx.x, b = blockIdx.y, t = blockIdx.z;
  const int tid = threadIdx.x;
  __shared__ float sc[S_];
  __shared__ float red[64];
  __shared__ float qs[HD_];
  qs[tid] = q[((size_t)t * B_ + b) * H_ + h * HD_ + tid];
  __syncthreads();
  // scores
  for (int s8 = 0; s8 < 8; ++s8) {
    int s = tid + s8 * 64;
    const bf16* kr = k + ((size_t)s * B_ + b) * H_ + h * HD_;
    float d = 0.0f;
    for (int e = 0; e < HD_; ++e) d += qs[e] * (float)kr[e];
    sc[s] = d;
  }
  __syncthreads();
  // max
  float mx = -1e30f;
  for (int s8 = 0; s8 < 8; ++s8) mx = fmaxf(mx, sc[tid + s8 * 64]);
  red[tid] = mx; __syncthreads();
  for (int st = 32; st > 0; st >>= 1) {
    if (tid < st) red[tid] = fmaxf(red[tid], red[tid + st]);
    __syncthreads();
  }
  mx = red[0]; __syncthreads();
  // exp + sum
  float sum = 0.0f;
  for (int s8 = 0; s8 < 8; ++s8) {
    int s = tid + s8 * 64;
    float e = __expf(sc[s] - mx);
    sc[s] = e; sum += e;
  }
  red[tid] = sum; __syncthreads();
  for (int st = 32; st > 0; st >>= 1) {
    if (tid < st) red[tid] += red[tid + st];
    __syncthreads();
  }
  const float inv = 1.0f / red[0];
  __syncthreads();
  // ctx[d=tid]
  float acc = 0.0f;
  for (int s = 0; s < S_; ++s)
    acc += sc[s] * (float)v[((size_t)s * B_ + b) * H_ + h * HD_ + tid];
  ctx[((size_t)t * B_ + b) * H_ + h * HD_ + tid] = (bf16)(acc * inv);
}

// t-major -> b-major permute (bf16 [4128,512])
__global__ void k_perm(const bf16* __restrict__ src, bf16* __restrict__ dst) {
  int i = blockIdx.x * 256 + threadIdx.x;
  if (i >= M1 * H_) return;
  int h = i % H_, row = i / H_;
  int t = row / B_, b = row % B_;
  dst[((size_t)b * T1_ + t) * H_ + h] = src[i];
}

// in-place log_softmax over rows of length `cols`
__global__ void __launch_bounds__(256) k_logsoftmax(float* __restrict__ out, int cols) {
  __shared__ float red[256];
  float* p = out + (size_t)blockIdx.x * cols;
  const int tid = threadIdx.x;
  float mx = -1e30f;
  for (int i = tid; i < cols; i += 256) mx = fmaxf(mx, p[i]);
  red[tid] = mx; __syncthreads();
  for (int st = 128; st > 0; st >>= 1) {
    if (tid < st) red[tid] = fmaxf(red[tid], red[tid + st]);
    __syncthreads();
  }
  mx = red[0]; __syncthreads();
  float sum = 0.0f;
  for (int i = tid; i < cols; i += 256) sum += __expf(p[i] - mx);
  red[tid] = sum; __syncthreads();
  for (int st = 128; st > 0; st >>= 1) {
    if (tid < st) red[tid] += red[tid + st];
    __syncthreads();
  }
  const float lse = mx + __logf(red[0]);
  for (int i = tid; i < cols; i += 256) p[i] -= lse;
}

// ---------------------------------------------------------------------------
static inline void gemm(hipStream_t st, const bf16* A, const bf16* W,
                        const float* bias, const float* Cadd,
                        float* Cf, bf16* Cbf, int M, int N, int K, float alpha) {
  dim3 grid((N + 63) / 64, (M / 16 + 3) / 4);
  k_gemm<<<grid, 128, 0, st>>>(A, W, bias, Cadd, Cf, Cbf, M, N, K, alpha);
}
static inline void cvt(hipStream_t st, const float* s, bf16* d, size_t n) {
  k_cvt<<<((int)n + 255) / 256, 256, 0, st>>>(s, d, (int)n);
}
static inline void zero32(hipStream_t st, void* p, size_t n32) {
  k_zero_u32<<<((int)n32 + 255) / 256, 256, 0, st>>>((unsigned*)p, (int)n32);
}

extern "C" void kernel_launch(void* const* d_in, const int* in_sizes, int n_in,
                              void* d_out, int out_size, void* d_ws, size_t ws_size,
                              hipStream_t stream) {
  const int*   inputs = (const int*)  d_in[0];
  const float* enc    = (const float*)d_in[1];
  const float* emb    = (const float*)d_in[2];
  const float* Wih    = (const float*)d_in[3];
  const float* Whh    = (const float*)d_in[4];
  const float* bih    = (const float*)d_in[5];
  const float* bhh    = (const float*)d_in[6];
  const float* inW    = (const float*)d_in[7];
  const float* inB    = (const float*)d_in[8];
  const float* outW   = (const float*)d_in[9];
  const float* outB   = (const float*)d_in[10];
  const float* projW  = (const float*)d_in[11];
  const float* genW   = (const float*)d_in[12];

  char* w = (char*)d_ws;
  size_t off = 0;
  auto take = [&](size_t bytes) -> char* {
    char* p = w + off;
    off = (off + bytes + 255) & ~(size_t)255;
    return p;
  };
  bf16*  x0    = (bf16*) take((size_t)M1 * E_ * 2);
  bf16*  seq1  = (bf16*) take((size_t)M1 * H_ * 2);
  bf16*  seq2  = (bf16*) take((size_t)M1 * H_ * 2);
  bf16*  WihB  = (bf16*) take((size_t)L_ * G4H * E_ * 2);
  bf16*  WhhB  = (bf16*) take((size_t)L_ * G4H * H_ * 2);
  bf16*  WinB  = (bf16*) take((size_t)3 * H_ * H_ * 2);
  bf16*  WoutB = (bf16*) take((size_t)H_ * H_ * 2);
  bf16*  WprojB= (bf16*) take((size_t)P_ * H_ * 2);
  bf16*  WgenB = (bf16*) take((size_t)V_ * P_ * 2);
  float* biasL = (float*)take((size_t)L_ * G4H * 4);
  float* pre   = (float*)take((size_t)M1 * G4H * 4);
  float* gbuf  = (float*)take((size_t)B_ * G4H * 4);
  float* cst   = (float*)take((size_t)B_ * H_ * 4);
  bf16*  hbf   = (bf16*) take((size_t)B_ * H_ * 2);
  float* qf    = (float*)take((size_t)M1 * H_ * 4);
  bf16*  kvb   = (bf16*) take((size_t)MS * H_ * 2);
  bf16*  kb    = (bf16*) take((size_t)MS * H_ * 2);
  bf16*  vb    = (bf16*) take((size_t)MS * H_ * 2);
  bf16*  ctxb  = (bf16*) take((size_t)M1 * H_ * 2);
  bf16*  outt  = (bf16*) take((size_t)M1 * H_ * 2);
  bf16*  outbm = (bf16*) take((size_t)M1 * H_ * 2);
  bf16*  hidb  = (bf16*) take((size_t)M1 * H_ * 2);
  (void)ws_size; (void)n_in; (void)in_sizes; (void)out_size;

  // weight conversion (once per launch; deterministic)
  cvt(stream, Wih,   WihB,  (size_t)L_ * G4H * E_);
  cvt(stream, Whh,   WhhB,  (size_t)L_ * G4H * H_);
  cvt(stream, inW,   WinB,  (size_t)3 * H_ * H_);
  cvt(stream, outW,  WoutB, (size_t)H_ * H_);
  cvt(stream, projW, WprojB,(size_t)P_ * H_);
  cvt(stream, genW,  WgenB, (size_t)V_ * P_);
  k_bias2<<<(L_ * G4H + 255) / 256, 256, 0, stream>>>(bih, bhh, biasL, L_ * G4H);

  // embedding (prepend SOS)
  k_embed<<<(M1 * E_ + 255) / 256, 256, 0, stream>>>(inputs, emb, x0);

  // -------- 2-layer LSTM --------
  for (int l = 0; l < L_; ++l) {
    const bf16* Xin    = (l == 0) ? x0   : seq1;
    bf16*       seqOut = (l == 0) ? seq1 : seq2;
    const bf16* Wihl = WihB + (size_t)l * G4H * E_;
    const bf16* Whhl = WhhB + (size_t)l * G4H * H_;
    // pregates[t,b,:] = x @ Wih^T + (bih+bhh)  — hoisted out of the time loop
    gemm(stream, Xin, Wihl, biasL + (size_t)l * G4H, nullptr,
         pre, nullptr, M1, G4H, 512, 1.0f);
    zero32(stream, cst, (size_t)B_ * H_);        // c = 0
    zero32(stream, hbf, (size_t)B_ * H_ / 2);    // h = 0 (bf16)
    for (int t = 0; t < T1_; ++t) {
      gemm(stream, hbf, Whhl, nullptr, pre + (size_t)t * B_ * G4H,
           gbuf, nullptr, B_, G4H, H_, 1.0f);
      k_lstm_cell<<<(B_ * H_ + 255) / 256, 256, 0, stream>>>(gbuf, cst, hbf, seqOut, t);
    }
  }

  // -------- cross attention --------
  // q = (seq @ Wq^T + bq) * HD^-0.5   (f32 out)
  gemm(stream, seq2, WinB, inB, nullptr, qf, nullptr, M1, H_, H_, 0.125f);
  k_prep_kv<<<(MS * H_ + 255) / 256, 256, 0, stream>>>(enc, kvb);
  gemm(stream, kvb, WinB + (size_t)H_ * H_,     inB + H_,     nullptr, nullptr, kb, MS, H_, H_, 1.0f);
  gemm(stream, kvb, WinB + (size_t)2 * H_ * H_, inB + 2 * H_, nullptr, nullptr, vb, MS, H_, H_, 1.0f);
  k_attn<<<dim3(NH_, B_, T1_), 64, 0, stream>>>(qf, kb, vb, ctxb);
  gemm(stream, ctxb, WoutB, outB, nullptr, nullptr, outt, M1, H_, H_, 1.0f);

  // -------- projections + log-softmax --------
  k_perm<<<(M1 * H_ + 255) / 256, 256, 0, stream>>>(outt, outbm);        // [B,T1,H] rows
  gemm(stream, outbm, WprojB, nullptr, nullptr, nullptr, hidb, M1, P_, H_, 1.0f);
  gemm(stream, hidb, WgenB, nullptr, nullptr, (float*)d_out, nullptr, M1, V_, P_, 1.0f);
  k_logsoftmax<<<M1, 256, 0, stream>>>((float*)d_out, V_);
}
